// Linear3Bit_34454227649199
// MI455X (gfx1250) — matrix operand order repaired and checked by reading, not compiled, not run on hardware
//
#include <hip/hip_runtime.h>
#include <stdint.h>

typedef _Float16 v16h __attribute__((ext_vector_type(16)));
typedef _Float16 v8h  __attribute__((ext_vector_type(8)));
typedef _Float16 v4h  __attribute__((ext_vector_type(4)));
typedef float    v8f  __attribute__((ext_vector_type(8)));

#define M_DIM 8192   // B*T = 4*2048
#define N_DIM 4096   // OUT
#define K_DIM 4096   // IN
#define BM 128
#define BN 128
#define BK 32
#define LDSA 40      // halves per LDS row (32 + 8 pad -> conflict-free b128 reads)
#define LDSB 40

__global__ __launch_bounds__(256)
void linear3bit_wmma_kernel(const float* __restrict__ x,
                            const uint8_t* __restrict__ wq,
                            const _Float16* __restrict__ scale,
                            const float* __restrict__ bias,
                            float* __restrict__ out)
{
    __shared__ _Float16 Al[BM * LDSA];   // x tile, f16, row-major [m][k]
    __shared__ _Float16 Bl[BN * LDSB];   // (q-3.5) tile, f16, row-major [n][k]

    const int tid  = threadIdx.x;
    const int lane = tid & 31;
    const int wave = tid >> 5;
    const int wm   = wave & 1;     // 0..1 -> 64 rows each
    const int wn   = wave >> 1;    // 0..3 -> 32 cols each

    const int mBase = blockIdx.y * BM;
    const int nBase = blockIdx.x * BN;

    const int l15 = lane & 15;
    const int lh  = lane >> 4;

    v8f acc[4][2] = {};

    // B staging assignment: 2 threads per weight row, 16 bytes each
    const int brow = tid >> 1;
    const int bko  = (tid & 1) * 16;
    const uint8_t* wq_row = wq + (size_t)(nBase + brow) * K_DIM + bko;

    for (int k0 = 0; k0 < K_DIM; k0 += BK) {
        // ---- stage A: 128x32 fp32 -> f16 in LDS (4 float4 per thread) ----
        #pragma unroll
        for (int i = 0; i < 4; ++i) {
            const int chunk = tid + i * 256;
            const int row   = chunk >> 3;
            const int c4    = chunk & 7;
            const float4 f = *(const float4*)(x + (size_t)(mBase + row) * K_DIM + k0 + c4 * 4);
            v4h h;
            h[0] = (_Float16)f.x; h[1] = (_Float16)f.y;
            h[2] = (_Float16)f.z; h[3] = (_Float16)f.w;
            *(v4h*)(&Al[row * LDSA + c4 * 4]) = h;
        }
        // ---- stage B: 128x32 u8 -> (q - 3.5) f16 in LDS (16 bytes per thread) ----
        {
            const uint4 u = *(const uint4*)(wq_row + k0);
            const uint32_t w[4] = {u.x, u.y, u.z, u.w};
            v8h lo, hi;
            #pragma unroll
            for (int j = 0; j < 8; ++j) {
                lo[j] = (_Float16)((float)((w[j >> 2]       >> ((j & 3) * 8)) & 0xffu) - 3.5f);
                hi[j] = (_Float16)((float)((w[(j + 8) >> 2] >> ((j & 3) * 8)) & 0xffu) - 3.5f);
            }
            *(v8h*)(&Bl[brow * LDSB + bko])     = lo;
            *(v8h*)(&Bl[brow * LDSB + bko + 8]) = hi;
        }
        // prefetch next k-tile while we compute this one
        if (k0 + BK < K_DIM) {
            __builtin_prefetch(x + (size_t)(mBase + (tid >> 3)) * K_DIM + (k0 + BK) + (tid & 7) * 4, 0, 1);
            __builtin_prefetch(wq_row + (k0 + BK), 0, 1);
        }
        __syncthreads();

        // ---- B fragments (32x16 f16, KxN): lane n=l15, k = lh*16 .. +16 contiguous ----
        v16h bfrag[2];
        #pragma unroll
        for (int nt = 0; nt < 2; ++nt) {
            const int nrow = wn * 32 + nt * 16 + l15;
            const v8h blo = *(const v8h*)(&Bl[nrow * LDSB + lh * 16]);
            const v8h bhi = *(const v8h*)(&Bl[nrow * LDSB + lh * 16 + 8]);
            #pragma unroll
            for (int i = 0; i < 8; ++i) { bfrag[nt][i] = blo[i]; bfrag[nt][i + 8] = bhi[i]; }
        }

        // ---- A fragments + 8 WMMAs ----
        #pragma unroll
        for (int mt = 0; mt < 4; ++mt) {
            const int arow = wm * 64 + mt * 16 + l15;
            const v8h alo = *(const v8h*)(&Al[arow * LDSA + lh * 16]);       // K = lh*16 .. +8
            const v8h ahi = *(const v8h*)(&Al[arow * LDSA + lh * 16 + 8]);   // K = lh*16+8 .. +8
            v16h afrag;
            #pragma unroll
            for (int i = 0; i < 8; ++i) { afrag[i] = alo[i]; afrag[i + 8] = ahi[i]; }

            acc[mt][0] = __builtin_amdgcn_wmma_f32_16x16x32_f16(
                false, afrag, false, bfrag[0], (short)0, acc[mt][0], false, false);
            acc[mt][1] = __builtin_amdgcn_wmma_f32_16x16x32_f16(
                false, afrag, false, bfrag[1], (short)0, acc[mt][1], false, false);
        }
        __syncthreads();
    }

    // ---- epilogue: out = acc * scale[n] + bias[n] (f32) ----
    #pragma unroll
    for (int nt = 0; nt < 2; ++nt) {
        const int n  = nBase + wn * 32 + nt * 16 + l15;
        const float s  = (float)scale[n];
        const float bv = bias[n];
        #pragma unroll
        for (int mt = 0; mt < 4; ++mt) {
            const int m0 = mBase + wm * 64 + mt * 16 + lh * 8;
            #pragma unroll
            for (int v = 0; v < 8; ++v) {
                out[(size_t)(m0 + v) * N_DIM + n] = acc[mt][nt][v] * s + bv;
            }
        }
    }
}

extern "C" void kernel_launch(void* const* d_in, const int* in_sizes, int n_in,
                              void* d_out, int out_size, void* d_ws, size_t ws_size,
                              hipStream_t stream) {
    const float*    x     = (const float*)d_in[0];
    const uint8_t*  wq    = (const uint8_t*)d_in[1];     // int8 codes 0..7
    const _Float16* scale = (const _Float16*)d_in[2];    // fp16 per-row scale
    const float*    bias  = (const float*)d_in[3];
    float*          out   = (float*)d_out;

    dim3 grid(N_DIM / BN, M_DIM / BM);   // 32 x 64
    linear3bit_wmma_kernel<<<grid, 256, 0, stream>>>(x, wq, scale, bias, out);
}
